// Tokenizer_55173149884874
// MI455X (gfx1250) — matrix operand order repaired and checked by reading, compiled, not run on hardware
//
#include <hip/hip_runtime.h>
#include <hip/hip_bf16.h>

typedef __attribute__((ext_vector_type(16))) _Float16 v16h;
typedef __attribute__((ext_vector_type(8)))  _Float16 v8h;
typedef __attribute__((ext_vector_type(8)))  float    v8f;

// ---------------------------------------------------------------------------
// Prep kernels
// ---------------------------------------------------------------------------

// im2col for 8x8 stride-8 patches: x [512,3,64,64] f32 -> A [32768,192] f16
// m = b*64 + oh*8 + ow ; kk = c*64 + kh*8 + kw
__global__ __launch_bounds__(256) void im2col_enc(const float* __restrict__ x,
                                                  _Float16* __restrict__ A) {
  int idx = blockIdx.x * 256 + threadIdx.x;
  if (idx >= 32768 * 192) return;
  int m  = idx / 192;
  int kk = idx - m * 192;
  int c  = kk >> 6, kh = (kk >> 3) & 7, kw = kk & 7;
  int b  = m >> 6,  oh = (m >> 3) & 7,  ow = m & 7;
  size_t src = ((size_t)(b * 3 + c) * 64 + (oh * 8 + kh)) * 64 + (ow * 8 + kw);
  A[idx] = (_Float16)x[src];
}

// in [R][C] f32 -> out [C][R] f16  (generic weight transpose+convert)
__global__ __launch_bounds__(256) void transpose_f32_to_f16(const float* __restrict__ in,
                                                            _Float16* __restrict__ out,
                                                            int R, int C) {
  int idx = blockIdx.x * 256 + threadIdx.x;
  if (idx >= R * C) return;
  int r = idx / C, c = idx - r * C;
  out[(size_t)c * R + r] = (_Float16)in[idx];
}

// dec_w [3,512,8,8] f32 -> B [512(k=z)][192(n=c*64+p)] f16
__global__ __launch_bounds__(256) void prep_dec(const float* __restrict__ dec_w,
                                                _Float16* __restrict__ B) {
  int idx = blockIdx.x * 256 + threadIdx.x;
  if (idx >= 512 * 192) return;
  int k = idx / 192, n = idx - k * 192;
  int c = n >> 6, p = n & 63;
  B[idx] = (_Float16)dec_w[((size_t)(c * 512 + k)) * 64 + p];
}

// codebook row norms: cnorm[n] = sum_e cb[n][e]^2
__global__ __launch_bounds__(256) void cnorm_kernel(const float* __restrict__ cb,
                                                    float* __restrict__ cnorm) {
  int n = blockIdx.x * 256 + threadIdx.x;
  if (n >= 1024) return;
  float s = 0.f;
  for (int e = 0; e < 512; ++e) { float v = cb[(size_t)n * 512 + e]; s += v * v; }
  cnorm[n] = s;
}

__global__ __launch_bounds__(256) void init_keys(unsigned long long* __restrict__ k) {
  k[blockIdx.x * 256 + threadIdx.x] = ~0ull;
}

// gather codebook rows by argmin token; write z_q f32 NCHW + f16 row-major copy
__global__ __launch_bounds__(256) void gather_zq(const unsigned long long* __restrict__ keys,
                                                 const float* __restrict__ cb,
                                                 float* __restrict__ zq_out,
                                                 _Float16* __restrict__ zq16) {
  int idx = blockIdx.x * 256 + threadIdx.x;      // 32768*512
  int m = idx >> 9, e = idx & 511;
  unsigned token = (unsigned)(keys[m] & 0xFFFFFFFFull);
  float v = cb[(size_t)token * 512 + e];
  int b = m >> 6, hw = m & 63;
  zq_out[((size_t)b * 512 + e) * 64 + hw] = v;
  zq16[(size_t)m * 512 + e] = (_Float16)v;
}

// ---------------------------------------------------------------------------
// Tiled WMMA GEMM: C[M,N] = A[M,K](f16) * B[K,N](f16), f32 accumulate.
// Block tile 128x64, BK=32, 256 threads = 8 waves (4x2), wave tile 32x32.
// Double-buffered software pipeline: tile t+1 staged (A via
// GLOBAL_LOAD_ASYNC_TO_LDS_B128, ASYNCcnt) while tile t runs on the WMMAs;
// drain with s_wait_asynccnt after compute, one barrier per K-step.
// MODE 0: ENC  -> out16 = relu(acc + bias[n])
// MODE 1: PRE  -> outf (z, NCHW scatter) and out16 = acc + bias[n]
// MODE 2: VQ   -> atomicMin u64 key (ordered(cnorm[n]-2acc) << 32 | n) per row
// MODE 3: POST -> out16 = acc + bias[n]
// MODE 4: DEC  -> outf (rec, transposed-conv NCHW scatter) = acc + bias[n>>6]
// ---------------------------------------------------------------------------
template <int MODE>
__global__ __launch_bounds__(256) void gemm_wmma(
    const _Float16* __restrict__ A, const _Float16* __restrict__ B,
    int M, int N, int K,
    const float* __restrict__ bias,
    _Float16* __restrict__ out16,
    float* __restrict__ outf,
    const float* __restrict__ cnorm,
    unsigned long long* __restrict__ argkey) {
  constexpr int LDA = 48;              // padded f16 stride (96B rows, 16B/32B aligned)
  constexpr int ASZ = 128 * LDA;       // f16 elems per A buffer
  constexpr int BSZ = 64 * LDA;        // f16 elems per B buffer
  __shared__ __align__(32) _Float16 As[2 * ASZ];
  __shared__ __align__(32) _Float16 Bs[2 * BSZ];

  const int tid  = threadIdx.x;
  const int lane = tid & 31;
  const int wave = tid >> 5;
  const int wm   = wave >> 1;   // 0..3
  const int wn   = wave & 1;    // 0..1
  const int half = lane >> 4;
  const int l16  = lane & 15;
  const int m0   = blockIdx.y * 128;
  const int n0   = blockIdx.x * 64;

  // LDS byte offset of As for the async-copy destination VGPR
  // (flat shared address low 32 bits == LDS offset per the aperture rules)
  const unsigned as_base = (unsigned)(unsigned long long)(uintptr_t)&As[0];

  // stage A tile 128x32 f16 via async copy: 2 x b128 per thread
  auto stageA = [&](int k0, int p) {
#pragma unroll
    for (int i = 0; i < 2; ++i) {
      int idx = tid + i * 256;          // 0..511 over 16B chunks
      int r = idx >> 2, c8 = idx & 3;
      const _Float16* src = A + (size_t)(m0 + r) * K + k0 + c8 * 8;
      unsigned dst = as_base + (unsigned)(p * ASZ + r * LDA + c8 * 8) * 2u;
      asm volatile("global_load_async_to_lds_b128 %0, %1, off"
                   :
                   : "v"(dst), "v"(src)
                   : "memory");
    }
  };
  // stage B tile transposed: Bs[n][k], 32x64 -> 8 f16 per thread
  auto stageB = [&](int k0, int p) {
    int r = tid >> 3, c8 = tid & 7;     // k row, n/8
    v8h v = *(const v8h*)(B + (size_t)(k0 + r) * N + n0 + c8 * 8);
    _Float16* bb = &Bs[p * BSZ];
#pragma unroll
    for (int j = 0; j < 8; ++j) bb[(c8 * 8 + j) * LDA + r] = v[j];
  };

  v8f acc[2][2] = {};

  // prologue: tile 0 into buffer 0
  stageA(0, 0);
  stageB(0, 0);
  asm volatile("s_wait_asynccnt 0x0" ::: "memory");
  __syncthreads();

  for (int k0 = 0; k0 < K; k0 += 32) {
    const int p = (k0 >> 5) & 1;
    const bool more = (k0 + 32 < K);

    // kick off tile t+1 into the other buffer (overlaps with WMMAs below)
    if (more) {
      stageA(k0 + 32, p ^ 1);
      stageB(k0 + 32, p ^ 1);
    }
    if (k0 + 64 < K) {  // keep L2 one more tile ahead of the async engine
      const _Float16* pf = A + (size_t)(m0 + (tid >> 1)) * K + (k0 + 64) + (tid & 1) * 16;
      __builtin_prefetch(pf, 0, 1);
    }

    // --- fragments in contiguous k order (k = 16*half + s) for A and B, from buffer p ---
    const _Float16* as = &As[p * ASZ];
    const _Float16* bs = &Bs[p * BSZ];
    v16h af[2], bf[2];
#pragma unroll
    for (int mi = 0; mi < 2; ++mi) {
      const _Float16* ap = &as[(wm * 32 + mi * 16 + l16) * LDA + half * 16];
      v8h lo = *(const v8h*)ap;
      v8h hi = *(const v8h*)(ap + 8);
#pragma unroll
      for (int e = 0; e < 8; ++e) { af[mi][e] = lo[e]; af[mi][8 + e] = hi[e]; }
    }
#pragma unroll
    for (int ni = 0; ni < 2; ++ni)
      bf[ni] = *(const v16h*)(&bs[(wn * 32 + ni * 16 + l16) * LDA + half * 16]);

#pragma unroll
    for (int mi = 0; mi < 2; ++mi)
#pragma unroll
      for (int ni = 0; ni < 2; ++ni)
        acc[mi][ni] = __builtin_amdgcn_wmma_f32_16x16x32_f16(
            false, af[mi], false, bf[ni], (short)0, acc[mi][ni], false, false);

    // drain this wave's async copies for tile t+1, publish via one barrier
    if (more) asm volatile("s_wait_asynccnt 0x0" ::: "memory");
    __syncthreads();
  }

  // --- epilogues ---
#pragma unroll
  for (int mi = 0; mi < 2; ++mi) {
    if (MODE == 2) {
      // VQ argmin: lane holds (m = base+8*half+v, n = n0+wn*32+ni*16+l16)
#pragma unroll
      for (int v = 0; v < 8; ++v) {
        int m = m0 + wm * 32 + mi * 16 + half * 8 + v;
        unsigned long long best = ~0ull;
#pragma unroll
        for (int ni = 0; ni < 2; ++ni) {
          int n = n0 + wn * 32 + ni * 16 + l16;
          float dist = cnorm[n] - 2.0f * acc[mi][ni][v];
          unsigned u = __float_as_uint(dist);
          u = (u & 0x80000000u) ? ~u : (u | 0x80000000u);  // order-preserving map
          unsigned long long key = ((unsigned long long)u << 32) | (unsigned)n;
          best = key < best ? key : best;
        }
        // same m lives in 16 lanes of the same half -> reduce, then 1 atomic
#pragma unroll
        for (int s = 1; s < 16; s <<= 1) {
          unsigned long long o = __shfl_xor(best, s, 32);
          best = o < best ? o : best;
        }
        if (l16 == 0) atomicMin(argkey + m, best);
      }
    } else {
#pragma unroll
      for (int ni = 0; ni < 2; ++ni) {
        int n = n0 + wn * 32 + ni * 16 + l16;
#pragma unroll
        for (int v = 0; v < 8; ++v) {
          int m = m0 + wm * 32 + mi * 16 + half * 8 + v;
          float val = acc[mi][ni][v];
          if (MODE == 0) {
            val += bias[n];
            val = val > 0.f ? val : 0.f;
            out16[(size_t)m * N + n] = (_Float16)val;
          } else if (MODE == 1) {
            val += bias[n];
            int b = m >> 6, hw = m & 63;
            outf[((size_t)b * N + n) * 64 + hw] = val;       // z, NCHW
            out16[(size_t)m * N + n] = (_Float16)val;        // feeds VQ GEMM
          } else if (MODE == 3) {
            val += bias[n];
            out16[(size_t)m * N + n] = (_Float16)val;        // feeds DEC GEMM
          } else {  // MODE == 4
            int c = n >> 6, p2 = n & 63;
            val += bias[c];
            int b = m >> 6, oh = (m >> 3) & 7, ow = m & 7;
            int kh = p2 >> 3, kw = p2 & 7;
            outf[(size_t)(b * 3 + c) * 4096 + (oh * 8 + kh) * 64 + (ow * 8 + kw)] = val;
          }
        }
      }
    }
  }
}

// ---------------------------------------------------------------------------
// Launch
// ---------------------------------------------------------------------------
extern "C" void kernel_launch(void* const* d_in, const int* in_sizes, int n_in,
                              void* d_out, int out_size, void* d_ws, size_t ws_size,
                              hipStream_t stream) {
  const float* x      = (const float*)d_in[0];
  const float* enc_w  = (const float*)d_in[1];
  const float* enc_b  = (const float*)d_in[2];
  const float* pre_w  = (const float*)d_in[3];
  const float* pre_b  = (const float*)d_in[4];
  const float* cb     = (const float*)d_in[5];
  const float* post_w = (const float*)d_in[6];
  const float* post_b = (const float*)d_in[7];
  const float* dec_w  = (const float*)d_in[8];
  const float* dec_b  = (const float*)d_in[9];
  (void)in_sizes; (void)n_in; (void)out_size; (void)ws_size;

  float* out     = (float*)d_out;
  float* z_out   = out;                 // [512,512,8,8]
  float* zq_out  = out + 16777216;      // [512,512,8,8]
  float* rec_out = out + 33554432;      // [512,3,64,64]

  char* ws = (char*)d_ws;
  size_t off = 0;
  auto alloc = [&](size_t bytes) {
    char* p = ws + off;
    off += (bytes + 255) & ~(size_t)255;
    return p;
  };
  _Float16* Aenc  = (_Float16*)alloc(32768ull * 192 * 2);
  _Float16* h1    = (_Float16*)alloc(32768ull * 512 * 2);
  _Float16* z16   = (_Float16*)alloc(32768ull * 512 * 2);
  _Float16* zq16  = (_Float16*)alloc(32768ull * 512 * 2);
  _Float16* Benc  = (_Float16*)alloc(192ull * 512 * 2);
  _Float16* Bpre  = (_Float16*)alloc(512ull * 512 * 2);
  _Float16* Bcode = (_Float16*)alloc(512ull * 1024 * 2);
  _Float16* Bpost = (_Float16*)alloc(512ull * 512 * 2);
  _Float16* Bdec  = (_Float16*)alloc(512ull * 192 * 2);
  float*    cnorm = (float*)alloc(1024 * 4);
  unsigned long long* argkey = (unsigned long long*)alloc(32768ull * 8);
  _Float16* d16 = h1;  // h1 is dead after PRE; reuse for post-quant activations

  // prep
  im2col_enc<<<24576, 256, 0, stream>>>(x, Aenc);
  transpose_f32_to_f16<<<384, 256, 0, stream>>>(enc_w, Benc, 512, 192);
  transpose_f32_to_f16<<<1024, 256, 0, stream>>>(pre_w, Bpre, 512, 512);
  transpose_f32_to_f16<<<2048, 256, 0, stream>>>(cb, Bcode, 1024, 512);
  transpose_f32_to_f16<<<1024, 256, 0, stream>>>(post_w, Bpost, 512, 512);
  prep_dec<<<384, 256, 0, stream>>>(dec_w, Bdec);
  cnorm_kernel<<<4, 256, 0, stream>>>(cb, cnorm);
  init_keys<<<128, 256, 0, stream>>>(argkey);

  // pipeline
  gemm_wmma<0><<<dim3(8, 256), 256, 0, stream>>>(Aenc, Benc, 32768, 512, 192,
                                                 enc_b, h1, nullptr, nullptr, nullptr);
  gemm_wmma<1><<<dim3(8, 256), 256, 0, stream>>>(h1, Bpre, 32768, 512, 512,
                                                 pre_b, z16, z_out, nullptr, nullptr);
  gemm_wmma<2><<<dim3(16, 256), 256, 0, stream>>>(z16, Bcode, 32768, 1024, 512,
                                                  nullptr, nullptr, nullptr, cnorm, argkey);
  gather_zq<<<65536, 256, 0, stream>>>(argkey, cb, zq_out, zq16);
  gemm_wmma<3><<<dim3(8, 256), 256, 0, stream>>>(zq16, Bpost, 32768, 512, 512,
                                                 post_b, d16, nullptr, nullptr, nullptr);
  gemm_wmma<4><<<dim3(3, 256), 256, 0, stream>>>(d16, Bdec, 32768, 192, 512,
                                                 dec_b, nullptr, rec_out, nullptr, nullptr);
}
